// SparseConv2d_33045478375971
// MI455X (gfx1250) — matrix operand order repaired and checked by reading, not compiled, not run on hardware
//
#include <hip/hip_runtime.h>

// ---------------------------------------------------------------------------
// SparseConv2d (masked 3x3 conv, stride 1, pad 1) as implicit GEMM on CDNA5
//   M = Cout = 256, N = B*H*W = 100352, K = Cin*9 = 2304  (~118 GFLOP)
// bf16 WMMA (v_wmma_f32_16x16x32_bf16), f32 accumulation.
// Double-buffered LDS pipeline; weight tiles staged with
// global_load_async_to_lds_b128 (ASYNCcnt), input gathers staged via regs.
// ---------------------------------------------------------------------------

typedef __attribute__((ext_vector_type(16))) __bf16 v16bf;
typedef __attribute__((ext_vector_type(8)))  __bf16 v8bf;
typedef __attribute__((ext_vector_type(8)))  float  v8f;

#define CIN    256
#define COUT   256
#define HH     56
#define WW     56
#define HW     (HH * WW)        // 3136
#define BATCH  32
#define NPIX   (BATCH * HW)     // 100352
#define MT     128              // Cout tile
#define NT     128              // pixel tile
#define KSTEP  32               // bf16 WMMA K
#define LDA    40               // padded LDS row (bf16 elems), 80 B, 16B aligned
#define LDB    40
#define NSTEPS 72               // 9 taps * (256/32) cin chunks

__device__ __forceinline__ void async_load_b128(unsigned lds_off, const void* g) {
    // gfx1250 async global->LDS, per-lane 16B, tracked by ASYNCcnt
    asm volatile("global_load_async_to_lds_b128 %0, %1, off"
                 :: "v"(lds_off), "v"(g) : "memory");
}
__device__ __forceinline__ void wait_async0() {
    asm volatile("s_wait_asynccnt 0x0" ::: "memory");
}

// ---- pre-pass: fuse weight*mask, convert to bf16, repack [tap][co][cin] ----
__global__ __launch_bounds__(256)
void prune_pack_kernel(const float* __restrict__ w, const float* __restrict__ m,
                       __bf16* __restrict__ wt) {
    int idx = blockIdx.x * 256 + threadIdx.x;      // over COUT*CIN*9
    if (idx >= COUT * CIN * 9) return;
    int t   = idx % 9;
    int cin = (idx / 9) % CIN;
    int co  = idx / (9 * CIN);
    float v = w[idx] * m[idx];
    wt[(t * COUT + co) * CIN + cin] = (__bf16)v;
}

// ---- main kernel: pipelined implicit-GEMM conv with bf16 WMMA ----
__global__ __launch_bounds__(256)
void sparse_conv_wmma_kernel(const float* __restrict__ inp,
                             const float* __restrict__ bias,
                             const __bf16* __restrict__ wt,
                             float* __restrict__ out) {
    __shared__ __bf16 lA[2][MT * LDA];   // [buf][co 0..127][k 0..31]
    __shared__ __bf16 lB[2][NT * LDB];   // [buf][pix 0..127][k 0..31]

    const int tid = threadIdx.x;
    const int nt  = blockIdx.x % (NPIX / NT);   // 784 pixel tiles
    const int mt  = blockIdx.x / (NPIX / NT);   // 2 cout tiles
    const int n0  = nt * NT;
    const int co0 = mt * MT;

    // B-fill coordinates: thread -> (pixel column, k-subrange)
    const int jcol = tid & 127;
    const int ksub = (tid >> 7) * 16;           // 0 or 16
    const int npix = n0 + jcol;
    const int bimg = npix / HW;
    const int rem  = npix % HW;
    const int oh   = rem / WW;
    const int ow   = rem % WW;
    const int gctr = bimg * (CIN * HW) + oh * WW + ow;  // center; add tap offs

    // A-fill coordinates: thread -> (row, 16-elem half) of 128x32 tile
    const int arow  = tid >> 1;
    const int ahalf = (tid & 1) * 16;

    // wave/lane mapping: 8 waves in 2(M) x 4(N), each wave 64x32 of output
    const int wv   = tid >> 5;
    const int lane = tid & 31;
    const int mwB  = (wv & 1) * 64;
    const int nwB  = (wv >> 1) * 32;
    const int lh   = lane >> 4;                 // K-half select
    const int ll   = lane & 15;                 // M/N index within 16

    v8f acc[4][2];
    #pragma unroll
    for (int i = 0; i < 4; ++i)
        #pragma unroll
        for (int j = 0; j < 2; ++j)
            #pragma unroll
            for (int r = 0; r < 8; ++r) acc[i][j][r] = 0.0f;

    float bregs[16];

    // async-stage one 128x32 bf16 weight tile into lA[buf]
    auto issueA = [&](int k, int buf) {
        const int tap = k >> 3;
        const int kc  = (k & 7) * KSTEP;
        const __bf16* src = wt + (tap * COUT + co0 + arow) * CIN + kc + ahalf;
        unsigned loff = (unsigned)(size_t)&lA[buf][arow * LDA + ahalf];
        async_load_b128(loff,      src);
        async_load_b128(loff + 16, src + 8);
    };
    // gather 16 input values (one pixel, 16 cin) into registers
    auto loadB = [&](int k) {
        const int tap = k >> 3;
        const int kc  = (k & 7) * KSTEP;
        const int kh  = tap / 3;
        const int kw  = tap - kh * 3;
        const int ih  = oh + kh - 1;
        const int iw  = ow + kw - 1;
        const bool valid = ((unsigned)ih < (unsigned)HH) &&
                           ((unsigned)iw < (unsigned)WW);
        const int gp = gctr + (kh - 1) * WW + (kw - 1);
        #pragma unroll
        for (int i = 0; i < 16; ++i) {
            const int cin = kc + ksub + i;
            bregs[i] = valid ? inp[gp + cin * HW] : 0.0f;
        }
    };
    // convert + write staged B values as two 16B LDS stores
    auto storeB = [&](int buf) {
        v8bf p0, p1;
        #pragma unroll
        for (int i = 0; i < 8; ++i) {
            p0[i] = (__bf16)bregs[i];
            p1[i] = (__bf16)bregs[i + 8];
        }
        *(v8bf*)&lB[buf][jcol * LDB + ksub]     = p0;
        *(v8bf*)&lB[buf][jcol * LDB + ksub + 8] = p1;
    };

    // ---- pipeline prologue: stage K-step 0 into buffer 0 ----
    issueA(0, 0);
    loadB(0);
    storeB(0);
    wait_async0();
    __syncthreads();

    // ---- main pipeline: one barrier per K-step ----
    for (int k = 0; k < NSTEPS; ++k) {
        const int cur = k & 1;
        const int nxt = cur ^ 1;

        if (k + 1 < NSTEPS) {
            issueA(k + 1, nxt);     // async weights -> LDS[nxt]
            loadB(k + 1);           // input gathers -> regs (overlap w/ WMMA)
        }

        // compute on LDS[cur]: 2 B-frags x 4 A-frags = 8 WMMAs per wave
        #pragma unroll
        for (int nf = 0; nf < 2; ++nf) {
            // B (K x N): lane ll = column; lanes 0-15 K{0..7,16..23}, lanes 16-31 K{8..15,24..31}
            const __bf16* bp = &lB[cur][(nwB + nf * 16 + ll) * LDB + lh * 8];
            v8bf b0 = *(const v8bf*)(bp);
            v8bf b1 = *(const v8bf*)(bp + 16);
            v16bf bb;
            #pragma unroll
            for (int i = 0; i < 8; ++i) { bb[i] = b0[i]; bb[i + 8] = b1[i]; }

            #pragma unroll
            for (int mf = 0; mf < 4; ++mf) {
                // A (M x K): lane ll = row; lanes 0-15 K{0..7,16..23},
                //            lanes 16-31 K{8..15,24..31}
                const __bf16* ap = &lA[cur][(mwB + mf * 16 + ll) * LDA + lh * 8];
                v8bf a0 = *(const v8bf*)(ap);
                v8bf a1 = *(const v8bf*)(ap + 16);
                v16bf aa;
                #pragma unroll
                for (int i = 0; i < 8; ++i) { aa[i] = a0[i]; aa[i + 8] = a1[i]; }

                acc[mf][nf] = __builtin_amdgcn_wmma_f32_16x16x32_bf16(
                    false, aa, false, bb, (short)0, acc[mf][nf], false, false);
            }
        }

        if (k + 1 < NSTEPS) storeB(nxt);   // regs -> LDS[nxt] (safe: nxt idle)
        wait_async0();                     // weight tile for nxt landed
        __syncthreads();
    }

    // ---- epilogue: add bias, scatter to NCHW ----
    // C/D layout: lane ll = column N, VGPR r -> row M = lh*8 + r
    #pragma unroll
    for (int nf = 0; nf < 2; ++nf) {
        const int n = n0 + nwB + nf * 16 + ll;
        const int b = n / HW;
        const int r = n % HW;
        const int obase = b * (COUT * HW) + r;          // + co*HW
        #pragma unroll
        for (int mf = 0; mf < 4; ++mf) {
            const int cobase = co0 + mwB + mf * 16 + lh * 8;
            #pragma unroll
            for (int rr = 0; rr < 8; ++rr) {
                const int co = cobase + rr;
                out[obase + co * HW] = acc[mf][nf][rr] + bias[co];
            }
        }
    }
}

extern "C" void kernel_launch(void* const* d_in, const int* in_sizes, int n_in,
                              void* d_out, int out_size, void* d_ws, size_t ws_size,
                              hipStream_t stream) {
    const float* inp    = (const float*)d_in[0];
    const float* weight = (const float*)d_in[1];
    const float* bias   = (const float*)d_in[2];
    const float* mask   = (const float*)d_in[3];
    __bf16* wt  = (__bf16*)d_ws;              // 256*256*9*2 B = 1.18 MB
    float*  out = (float*)d_out;

    const int wtotal = COUT * CIN * 9;
    hipLaunchKernelGGL(prune_pack_kernel,
                       dim3((wtotal + 255) / 256), dim3(256), 0, stream,
                       weight, mask, wt);

    hipLaunchKernelGGL(sparse_conv_wmma_kernel,
                       dim3((NPIX / NT) * (COUT / MT)), dim3(256), 0, stream,
                       inp, bias, wt, out);
}
